// AttentionMLP_8151847928386
// MI455X (gfx1250) — compile-verified, matrix operand order checked by reading, not run on hardware
//
#include <hip/hip_runtime.h>
#include <hip/hip_bf16.h>

// ---------------------------------------------------------------------------
// AttentionMLP, algebraically reduced:
//   scores[b,t] = x[b,t,:] . (W @ v)   (+ per-batch constant, killed by softmax)
//   at = softmax_t(scores);  out[b,:] = sum_t at[b,t] * x[b,t,:]
// g, Wg, b are provably irrelevant to the output (softmax shift invariance).
// Bandwidth-bound: 1 HBM pass over x (134 MB ~5.8us @ 23.3 TB/s) + 1 L2-resident
// pass (x fits in the 192 MB global L2). Compute after reduction is ~134 MFLOP,
// i.e. negligible vs any matrix-pipe ceiling -> data movement is everything.
// Score pass uses V_WMMA_F32_16X16X4_F32 (exact f32 matrix pipe).
// ---------------------------------------------------------------------------

typedef float     v2f  __attribute__((ext_vector_type(2)));
typedef float     v8f  __attribute__((ext_vector_type(8)));
typedef _Float16  v16h __attribute__((ext_vector_type(16)));

#define NB 64
#define NT 2048
#define ND 256   // == U == Dg

// ---------------- Kernel A: wv[d] = sum_u W[d,u] * v[u] ---------------------
__global__ __launch_bounds__(ND) void k_wv(const float* __restrict__ W,
                                           const float* __restrict__ v,
                                           float* __restrict__ wv) {
    __shared__ float sv[ND];
    sv[threadIdx.x] = v[threadIdx.x];
    __syncthreads();
    const float* row = W + (size_t)threadIdx.x * ND;
    float acc = 0.f;
#pragma unroll 8
    for (int u = 0; u < ND; ++u) acc = fmaf(row[u], sv[u], acc);
    wv[threadIdx.x] = acc;
}

// ---------------- Kernel B: scores[r] = x[r,:] . wv  via WMMA ---------------
// One wave per 16-row tile of the flattened (B*T, D) matrix.
// 8 waves / 256-thread block -> 128 rows per block -> 1024 blocks total.
// A-matrix f32 16x4 layout: lane L<16 holds row L, K pair {0,1}; lane L>=16
// holds row L-16, K pair {2,3} -> one 8B load per lane per WMMA step.
__global__ __launch_bounds__(256) void k_scores(const float* __restrict__ x,
                                                const float* __restrict__ wv,
                                                float* __restrict__ scores) {
    __shared__ float swv[ND];
    swv[threadIdx.x] = wv[threadIdx.x];
    __syncthreads();

    const unsigned lane = threadIdx.x & 31u;          // wave32
    const unsigned wave = threadIdx.x >> 5;           // 0..7
    const unsigned tile = blockIdx.x * 8u + wave;     // 16-row tile id
    const unsigned row0 = tile * 16u;
    const unsigned m    = lane & 15u;                 // row within tile
    const unsigned h    = lane >> 4;                  // lane half (0/1)
    const float* rowbase = x + (size_t)(row0 + m) * ND;

    v8f acc = {};

#if __has_builtin(__builtin_amdgcn_wmma_f32_16x16x4_f32)
    const unsigned koff = h * 2u;                     // 0 or 2
#pragma unroll 8
    for (int k0 = 0; k0 < ND; k0 += 4) {
        v2f a = *(const v2f*)(rowbase + k0 + koff);
        v2f b;                                        // wv chunk, bcast cols
        b.x = swv[k0 + koff + 0];
        b.y = swv[k0 + koff + 1];
        acc = __builtin_amdgcn_wmma_f32_16x16x4_f32(
                  false, a, false, b, (short)0, acc, false, false);
    }
#else
    // Fallback: f16 WMMA 16x16x32 (x/wv converted in-register).
    for (int k0 = 0; k0 < ND; k0 += 32) {
        v16h a, bb;
#pragma unroll
        for (int e = 0; e < 16; ++e) {
            const int ka = k0 + ((e < 8) ? (int)(h * 8u) + e
                                         : 16 + (int)(h * 8u) + (e - 8));
            a[e]  = (_Float16)rowbase[ka];
            bb[e] = (_Float16)swv[k0 + (int)(h * 16u) + e];
        }
        acc = __builtin_amdgcn_wmma_f32_16x16x32_f16(
                  false, a, false, bb, (short)0, acc, false, false);
    }
#endif

    // D layout: VGPR r -> M=r (lanes 0-15) / M=8+r (lanes 16-31); all N equal.
    if (lane == 0) {
#pragma unroll
        for (int r = 0; r < 8; ++r) scores[row0 + r] = acc[r];
    } else if (lane == 16) {
#pragma unroll
        for (int r = 0; r < 8; ++r) scores[row0 + 8 + r] = acc[r];
    }
}

// -------- Kernel C: per-batch softmax + weighted sum (L2-resident pass) -----
// 1024 threads = 32 waves = a full WGP per batch. T is split across 4 groups
// of 256 threads (group g owns t in [g*512, g*512+512)); each thread keeps 4
// independent accumulators to break the FMA dependency chain; group partials
// are combined through LDS.
__global__ __launch_bounds__(1024) void k_softmax_reduce(const float* __restrict__ x,
                                                         const float* __restrict__ scores,
                                                         float* __restrict__ out) {
    __shared__ float p[NT];        // 8 KB probabilities
    __shared__ float red[1024];    // 4 KB reductions / group partials
    const int b   = blockIdx.x;
    const int tid = threadIdx.x;   // 0..1023
    const float* sb = scores + (size_t)b * NT;

    // ---- softmax over 2048 scores ----
    float mx = -INFINITY;
    for (int t = tid; t < NT; t += 1024) { float s = sb[t]; p[t] = s; mx = fmaxf(mx, s); }
    red[tid] = mx; __syncthreads();
    for (int off = 512; off > 0; off >>= 1) {
        if (tid < off) red[tid] = fmaxf(red[tid], red[tid + off]);
        __syncthreads();
    }
    mx = red[0]; __syncthreads();

    float sum = 0.f;
    for (int t = tid; t < NT; t += 1024) { float e = __expf(p[t] - mx); p[t] = e; sum += e; }
    red[tid] = sum; __syncthreads();
    for (int off = 512; off > 0; off >>= 1) {
        if (tid < off) red[tid] += red[tid + off];
        __syncthreads();
    }
    const float inv = 1.0f / red[0];
    __syncthreads();
    for (int t = tid; t < NT; t += 1024) p[t] *= inv;
    __syncthreads();

    // ---- weighted sum: out[b,d] = sum_t p[t] * x[b,t,d] ----
    const int g = tid >> 8;        // t-group 0..3
    const int d = tid & 255;       // column
    const int t0 = g * (NT / 4);   // 512 t's per group
    const float* xb = x + ((size_t)b * NT + t0) * ND + d;   // coalesced: lane=d
    float a0 = 0.f, a1 = 0.f, a2 = 0.f, a3 = 0.f;
    for (int t = 0; t < NT / 4; t += 4) {
        a0 = fmaf(p[t0 + t + 0], xb[(size_t)(t + 0) * ND], a0);
        a1 = fmaf(p[t0 + t + 1], xb[(size_t)(t + 1) * ND], a1);
        a2 = fmaf(p[t0 + t + 2], xb[(size_t)(t + 2) * ND], a2);
        a3 = fmaf(p[t0 + t + 3], xb[(size_t)(t + 3) * ND], a3);
    }
    red[tid] = (a0 + a1) + (a2 + a3);
    __syncthreads();
    if (tid < ND)
        out[b * ND + tid] = (red[tid] + red[tid + 256]) + (red[tid + 512] + red[tid + 768]);
}

// ---------------------------------------------------------------------------
extern "C" void kernel_launch(void* const* d_in, const int* in_sizes, int n_in,
                              void* d_out, int out_size, void* d_ws, size_t ws_size,
                              hipStream_t stream) {
    const float* x = (const float*)d_in[0];
    // d_in[1]=g, d_in[3]=Wg, d_in[4]=b: mathematically irrelevant (see header).
    const float* W = (const float*)d_in[2];
    const float* v = (const float*)d_in[5];
    float* out = (float*)d_out;

    // Workspace layout: [wv: 256 f32][scores: 64*2048 f32] = ~525 KB.
    float* wv     = (float*)d_ws;
    float* scores = wv + ND;

    k_wv<<<1, ND, 0, stream>>>(W, v, wv);
    k_scores<<<(NB * NT) / 128, 256, 0, stream>>>(x, wv, scores);
    k_softmax_reduce<<<NB, 1024, 0, stream>>>(x, scores, out);
}
